// UntiedGalLstm_54649163874591
// MI455X (gfx1250) — matrix operand order repaired and checked by reading, not compiled, not run on hardware
//
#include <hip/hip_runtime.h>
#include <stdint.h>

// ---------------- problem constants ----------------
#define LSTM_B   4096
#define LSTM_D   2048
#define LSTM_H   2048
#define LSTM_4H  8192

// ---------------- tiling ----------------
#define BM   128   // batch rows per block
#define BNC  32    // c/h (hidden) columns per block -> 128 gate columns
#define BGN  128   // gate columns per block = 4*BNC
#define KB   32    // K step (bf16 WMMA K)
#define LDK  40    // padded LDS K-stride (bf16 elems): 80B rows, 16B aligned

typedef __bf16 bf16;
typedef __attribute__((ext_vector_type(16))) __bf16 v16bf;
typedef __attribute__((ext_vector_type(8)))  __bf16 v8bf;
typedef __attribute__((ext_vector_type(4)))  __bf16 v4bf;
typedef __attribute__((ext_vector_type(8)))  float  v8f;

// ---------- fast transcendental helpers (native CDNA5 TRANS ops) ----------
__device__ __forceinline__ float fast_exp2(float x) {
#if __has_builtin(__builtin_amdgcn_exp2f)
    return __builtin_amdgcn_exp2f(x);          // v_exp_f32
#else
    return exp2f(x);
#endif
}

__device__ __forceinline__ float fast_rcp(float x) {
#if __has_builtin(__builtin_amdgcn_rcpf)
    return __builtin_amdgcn_rcpf(x);           // v_rcp_f32
#else
    return 1.0f / x;
#endif
}

__device__ __forceinline__ float sigmoid_f(float x) {
    // 1 / (1 + 2^(-x*log2(e)))
    return fast_rcp(1.0f + fast_exp2(-1.4426950408889634f * x));
}

__device__ __forceinline__ float tanh_f(float x) {
#if __has_builtin(__builtin_amdgcn_tanhf)
    return __builtin_amdgcn_tanhf(x);          // v_tanh_f32
#elif __has_builtin(__builtin_amdgcn_tanh_f32)
    return __builtin_amdgcn_tanh_f32(x);
#else
    // tanh(x) = 1 - 2/(exp2(2x*log2(e)) + 1); saturates correctly at +/-inf
    return 1.0f - 2.0f * fast_rcp(1.0f + fast_exp2(2.8853900817779268f * x));
#endif
}

__device__ __forceinline__ void wait_asynccnt0() {
#if __has_builtin(__builtin_amdgcn_s_wait_asynccnt)
    __builtin_amdgcn_s_wait_asynccnt(0);
#else
    asm volatile("s_wait_asynccnt 0" ::: "memory");
#endif
}

__global__ __launch_bounds__(256)
void lstm_wmma_fused(const float* __restrict__ X,
                     const float* __restrict__ Ht,
                     const float* __restrict__ Ct,
                     const float* __restrict__ Wi,
                     const float* __restrict__ Wh,
                     const float* __restrict__ bi,
                     const float* __restrict__ bh,
                     float* __restrict__ Hout,
                     float* __restrict__ Cout)
{
    __shared__ bf16  As[BM  * LDK];    // A tile: [row][k], bf16, row-major
    __shared__ bf16  Bs[BGN * LDK];    // B tile: [gate-col][k], bf16 (transposed on store)
    __shared__ float Cs[BM * BNC];     // c_t tile, filled by async-to-LDS during phase 1

    const int tid  = threadIdx.x;
    const int lane = tid & 31;
    const int wave = tid >> 5;                 // 0..7
    const int colBase = blockIdx.x * BNC;      // hidden-column base
    const int row0    = blockIdx.y * BM;       // batch-row base

    v8f acc[8] = {};                           // 8 col-tiles of 16x16 f32

    // B staging assignment: thread -> (gate col, K half)
    const int bgc  = tid & 127;                      // gate col 0..127
    const int bkh  = (tid >> 7) * 16;                // K offset 0 or 16
    const int wcol = (bgc >> 5) * LSTM_H + colBase + (bgc & 31); // col in W[*,8192]

    // Fragment addressing (per ISA 16-bit layouts, wave32)
    const int mrow  = wave * 16 + (lane & 15);       // A row this lane reads
    const int akoff = (lane < 16) ? 0 : 8;           // A: K {akoff..+7, akoff+16..+23}
    const int bkoff = (lane < 16) ? 0 : 8;           // B: K {bkoff..+7, bkoff+16..+23}
    const int gcbase = lane & 15;

    #pragma unroll 1
    for (int phase = 0; phase < 2; ++phase) {
        const float* __restrict__ Asrc = phase ? Ht : X;   // both [B,2048]
        const float* __restrict__ Wsrc = phase ? Wh : Wi;  // both [2048,8192]

        // ---- at start of phase 1, kick off async c_t tile -> LDS ----
        // Outside the K-loop: address regs die immediately after issue and
        // the whole H@Wh GEMM (64 K-steps) hides the latency. Consumed in
        // the epilogue after s_wait_asynccnt 0.
        if (phase == 1) {
            #pragma unroll
            for (int i = 0; i < 4; ++i) {
                int q  = i * 256 + tid;      // 1024 float4 slots
                int r  = q >> 3;             // row 0..127
                int c4 = q & 7;              // float4 col index
                uint32_t ldst = (uint32_t)(uintptr_t)&Cs[r * BNC + c4 * 4];
                const float* gsrc = Ct + (row0 + r) * LSTM_H + colBase + c4 * 4;
                asm volatile("global_load_async_to_lds_b128 %0, %1, off"
                             :: "v"(ldst), "v"(gsrc) : "memory");
            }
        }

        #pragma unroll 1
        for (int k0 = 0; k0 < LSTM_D; k0 += KB) {
            // ---------- global fetch (fp32) ----------
            float4 av[4];
            #pragma unroll
            for (int i = 0; i < 4; ++i) {
                int q  = i * 256 + tid;        // 0..1023 float4 slots
                int r  = q >> 3;               // row 0..127
                int c4 = q & 7;                // float4 index along K
                av[i] = *(const float4*)(Asrc + (row0 + r) * LSTM_D + k0 + c4 * 4);
            }
            float bv[16];
            #pragma unroll
            for (int j = 0; j < 16; ++j)
                bv[j] = Wsrc[(k0 + bkh + j) * LSTM_4H + wcol];

            // prefetch next K-step's weight rows (global_prefetch_b8)
            __builtin_prefetch(Wsrc + (((k0 + KB) & (LSTM_D - 1)) + bkh) * LSTM_4H + wcol, 0, 3);

            __syncthreads();   // previous iteration's compute done reading LDS

            // ---------- convert fp32 -> bf16, stage in LDS ----------
            #pragma unroll
            for (int i = 0; i < 4; ++i) {
                int q  = i * 256 + tid;
                int r  = q >> 3;
                int c4 = q & 7;
                v4bf t;
                t[0] = (bf16)av[i].x; t[1] = (bf16)av[i].y;
                t[2] = (bf16)av[i].z; t[3] = (bf16)av[i].w;
                *(v4bf*)&As[r * LDK + c4 * 4] = t;           // 8B store
            }
            v8bf b0, b1;
            #pragma unroll
            for (int j = 0; j < 8; ++j) { b0[j] = (bf16)bv[j]; b1[j] = (bf16)bv[j + 8]; }
            *(v8bf*)&Bs[bgc * LDK + bkh    ] = b0;           // 16B stores, transposed
            *(v8bf*)&Bs[bgc * LDK + bkh + 8] = b1;

            __syncthreads();

            // ---------- WMMA compute ----------
            // A fragment: lane holds row mrow, K = {akoff..akoff+7, akoff+16..akoff+23}
            v8bf alo = *(const v8bf*)&As[mrow * LDK + akoff];
            v8bf ahi = *(const v8bf*)&As[mrow * LDK + akoff + 16];
            v16bf afrag = __builtin_shufflevector(alo, ahi,
                0, 1, 2, 3, 4, 5, 6, 7, 8, 9, 10, 11, 12, 13, 14, 15);

            // B fragments: software-pipelined so tile t+1's ds_loads are in
            // flight while WMMA t executes (keeps s_wait_dscnt >= 1).
            auto load_bfrag = [&](int t) -> v16bf {
                int gc = t * 16 + gcbase;
                v8bf blo = *(const v8bf*)&Bs[gc * LDK + bkoff];
                v8bf bhi = *(const v8bf*)&Bs[gc * LDK + bkoff + 16];
                return __builtin_shufflevector(blo, bhi,
                    0, 1, 2, 3, 4, 5, 6, 7, 8, 9, 10, 11, 12, 13, 14, 15);
            };

            v16bf bcur = load_bfrag(0);
            #pragma unroll
            for (int t = 0; t < 8; ++t) {
                v16bf bnext = bcur;
                if (t < 7) bnext = load_bfrag(t + 1);   // constant-folded guard
                acc[t] = __builtin_amdgcn_wmma_f32_16x16x32_bf16(
                    /*neg_a=*/false, afrag, /*neg_b=*/false, bcur,
                    /*c_mod=*/(short)0, acc[t],
                    /*reuse_a=*/false, /*reuse_b=*/false);
                bcur = bnext;
            }
        }
    }

    // ---------- bias loads issued before the async wait (latency overlap) ----------
    const int n       = lane & 15;
    const int halfsel = lane >> 4;
    float bsum[2][4];
    #pragma unroll
    for (int chunk = 0; chunk < 2; ++chunk) {
        const int hcol = colBase + chunk * 16 + n;
        #pragma unroll
        for (int g = 0; g < 4; ++g)
            bsum[chunk][g] = bi[g * LSTM_H + hcol] + bh[g * LSTM_H + hcol];
    }

    // ---------- wait for async c_t tile, then sync all waves ----------
    wait_asynccnt0();
    __syncthreads();

    // ---------- fused LSTM epilogue ----------
    // C/D layout: element e of lane l -> row (e + (l>>4)*8), col (l&15)
    #pragma unroll
    for (int chunk = 0; chunk < 2; ++chunk) {
        const int c    = chunk * 16 + n;       // 0..31 within block
        const int hcol = colBase + c;          // global hidden column

        #pragma unroll
        for (int e = 0; e < 8; ++e) {
            const int r    = wave * 16 + e + halfsel * 8;   // row within tile
            const int grow = row0 + r;
            const int idx  = grow * LSTM_H + hcol;

            const float gi_ = acc[0 + chunk][e] + bsum[chunk][0];
            const float gf_ = acc[2 + chunk][e] + bsum[chunk][1];
            const float gg_ = acc[4 + chunk][e] + bsum[chunk][2];
            const float go_ = acc[6 + chunk][e] + bsum[chunk][3];

            const float iv = sigmoid_f(gi_);
            const float fv = sigmoid_f(gf_);
            const float gv = tanh_f(gg_);
            const float ov = sigmoid_f(go_);

            const float cold = Cs[r * BNC + c];             // from async LDS tile
            const float cnew = fv * cold + iv * gv;
            const float hnew = ov * tanh_f(cnew);

            Cout[idx] = cnew;
            Hout[idx] = hnew;
        }
    }
}

extern "C" void kernel_launch(void* const* d_in, const int* in_sizes, int n_in,
                              void* d_out, int out_size, void* d_ws, size_t ws_size,
                              hipStream_t stream) {
    const float* X  = (const float*)d_in[0];   // input  [B, D]
    const float* Ht = (const float*)d_in[1];   // h_t    [B, H]
    const float* Ct = (const float*)d_in[2];   // c_t    [B, H]
    const float* Wi = (const float*)d_in[3];   // w_i    [D, 4H]
    const float* Wh = (const float*)d_in[4];   // w_h    [H, 4H]
    const float* bi = (const float*)d_in[5];   // b_i    [4H]
    const float* bh = (const float*)d_in[6];   // b_h    [4H]

    float* Hout = (float*)d_out;                               // h_new [B, H]
    float* Cout = (float*)d_out + (size_t)LSTM_B * LSTM_H;     // c_new [B, H]

    dim3 grid(LSTM_H / BNC, LSTM_B / BM);      // 64 x 32
    dim3 block(256, 1, 1);
    lstm_wmma_fused<<<grid, block, 0, stream>>>(X, Ht, Ct, Wi, Wh, bi, bh, Hout, Cout);
}
